// LlamaAttention_49392123904401
// MI455X (gfx1250) — matrix operand order repaired and checked by reading, not compiled, not run on hardware
//
#include <hip/hip_runtime.h>

// ---------------------------------------------------------------------------
// LlamaAttention (position-local head-mixing variant) for MI455X / gfx1250.
// bf16 WMMA (v_wmma_f32_16x16x32_bf16) with f32 accumulation.
//
// Shapes: B=4, S=8192 -> M = 32768 tokens, HIDDEN = 1024, 16 heads x 64 dim.
//
// ~277 GFLOP total, ~0.8 GB HBM traffic with bf16 staging -> compute-bound at
// the bf16 matrix rate.  GEMMs use 2M x 2N register blocking: two A fragments
// (LDS) x two B fragments (L2-resident packed weights) feed four independent
// WMMA accumulation chains => 2 x 128-bit loads per WMMA.
// ---------------------------------------------------------------------------

typedef __attribute__((ext_vector_type(16))) __bf16        bf16x16;
typedef __attribute__((ext_vector_type(8)))  float         f32x8;
typedef __attribute__((ext_vector_type(4)))  float         f32x4;
typedef __attribute__((ext_vector_type(4)))  unsigned int  u32x4;
typedef __attribute__((ext_vector_type(2)))  unsigned int  u32x2;

#define HID   1024
#define NTOK  32768
#define KTILES 32            // 1024 / 32
#define LDSTRIDE 1040        // padded row stride (elements): 2080 B = 8-bank skew

union BF16Frag { bf16x16 v; u32x4 u[2]; };

__device__ __forceinline__ bf16x16 load_frag_lds(const __bf16* p0, const __bf16* p1) {
    BF16Frag f;
    f.u[0] = *(const u32x4*)p0;
    f.u[1] = *(const u32x4*)p1;
    return f.v;
}

__device__ __forceinline__ bf16x16 load_frag_glb(const __bf16* p) {
    BF16Frag f;
    f.u[0] = ((const u32x4*)p)[0];
    f.u[1] = ((const u32x4*)p)[1];
    return f.v;
}

// ---------------------------------------------------------------------------
// Kernel 1: convert fp32 weights -> bf16, pre-swizzled into WMMA B-fragment
// order.  Fragment (nT,kT) of weight wi covers Y[:,nT*16..+16), K kT*32..+32.
// Lane l holds column n = nT*16 + (l&15), K = (l>>4)*8 + j + (j&8) (j = 0..15),
// 16 contiguous bf16 per lane:
//   packed[wi][((nT*32 + kT)*32 + lane)*16 + j] = bf16( W[n*1024 + k] )
// ---------------------------------------------------------------------------
__global__ __launch_bounds__(256) void pack_w(const float* __restrict__ W0,
                                              const float* __restrict__ W1,
                                              const float* __restrict__ W2,
                                              const float* __restrict__ W3,
                                              __bf16* __restrict__ out) {
    const int tid  = threadIdx.x;
    const int frag = blockIdx.x * 8 + (tid >> 5);   // 8192 fragments total
    const int lane = tid & 31;
    const int wi   = frag >> 11;                    // 2048 fragments / weight
    const int rem  = frag & 2047;
    const float* W = (wi == 0) ? W0 : (wi == 1) ? W1 : (wi == 2) ? W2 : W3;
    const int nT = rem >> 5, kT = rem & 31;
    const int n     = nT * 16 + (lane & 15);
    const int kbase = kT * 32 + (lane >> 4) * 8;
    __bf16* o = out + (size_t)wi * (HID * HID) + (((size_t)rem * 32 + lane) * 16);
#pragma unroll
    for (int j = 0; j < 16; ++j)
        o[j] = (__bf16)W[(size_t)n * HID + kbase + j + (j & 8)];
}

// ---------------------------------------------------------------------------
// Kernel 2: fused Q/K/V projection.  1 block = 32 tokens (2 M-subtiles),
// 8 waves.  Wave w owns 24 of the 192 (3 weights x 64) N-tiles, processed as
// 12 pairs; with 2 M-subtiles that is 4 accumulator chains per iteration.
// ---------------------------------------------------------------------------
__global__ __launch_bounds__(256, 1) void qkv_gemm(const float* __restrict__ X,
                                                   const __bf16* __restrict__ Wp,
                                                   __bf16* __restrict__ Qo,
                                                   __bf16* __restrict__ Ko,
                                                   __bf16* __restrict__ Vo) {
    __shared__ __bf16 sA[32 * LDSTRIDE];           // 32 tokens x 1024 bf16, padded
    const int tid = threadIdx.x;
    const size_t m0 = (size_t)blockIdx.x * 32;

    // Cooperative load + fp32->bf16 convert of the X tile (8192 float4 chunks).
#pragma unroll
    for (int i = 0; i < 32; ++i) {
        const int c = tid + i * 256;
        const int e = c * 4, row = e >> 10, col = e & 1023;
        const f32x4 f = *(const f32x4*)(X + (m0 + row) * HID + col);
        union { __bf16 h[4]; u32x2 u; } pk;
        pk.h[0] = (__bf16)f.x; pk.h[1] = (__bf16)f.y;
        pk.h[2] = (__bf16)f.z; pk.h[3] = (__bf16)f.w;
        *(u32x2*)&sA[row * LDSTRIDE + col] = pk.u;
    }
    __syncthreads();

    const int w = tid >> 5, lane = tid & 31;
    const int lm = lane & 15, lc = lane >> 4;

    for (int i = 0; i < 12; ++i) {
        const int t0 = w * 24 + i;                  // pair of N-tiles
        const int t1 = t0 + 12;
        const int wsel0 = t0 >> 6, nT0 = t0 & 63;   // 0:Wq 1:Wk 2:Wv
        const int wsel1 = t1 >> 6, nT1 = t1 & 63;
        const __bf16* wp0 = Wp + (size_t)wsel0 * (HID * HID)
                               + ((size_t)nT0 * 32 * 32 + lane) * 16;
        const __bf16* wp1 = Wp + (size_t)wsel1 * (HID * HID)
                               + ((size_t)nT1 * 32 * 32 + lane) * 16;
        f32x8 acc00 = {}, acc01 = {}, acc10 = {}, acc11 = {};
#pragma unroll 2
        for (int kT = 0; kT < KTILES; ++kT) {
            // A fragments (ISA 16-bit A 16x32 layout): lane row lm,
            // chunk0 K = kT*32 + lc*8, chunk1 K = chunk0 + 16.
            const int ab0 = lm * LDSTRIDE + kT * 32 + lc * 8;
            const int ab1 = (lm + 16) * LDSTRIDE + kT * 32 + lc * 8;
            const bf16x16 a0 = load_frag_lds(&sA[ab0], &sA[ab0 + 16]);
            const bf16x16 a1 = load_frag_lds(&sA[ab1], &sA[ab1 + 16]);
            const bf16x16 b0 = load_frag_glb(wp0 + (size_t)kT * 512);
            const bf16x16 b1 = load_frag_glb(wp1 + (size_t)kT * 512);
            __builtin_prefetch(wp0 + (size_t)(kT + 4) * 512, 0, 3);
            __builtin_prefetch(wp1 + (size_t)(kT + 4) * 512, 0, 3);
            acc00 = __builtin_amdgcn_wmma_f32_16x16x32_bf16(
                false, a0, false, b0, (short)0, acc00, false, false);
            acc01 = __builtin_amdgcn_wmma_f32_16x16x32_bf16(
                false, a0, false, b1, (short)0, acc01, false, false);
            acc10 = __builtin_amdgcn_wmma_f32_16x16x32_bf16(
                false, a1, false, b0, (short)0, acc10, false, false);
            acc11 = __builtin_amdgcn_wmma_f32_16x16x32_bf16(
                false, a1, false, b1, (short)0, acc11, false, false);
        }
        __bf16* outp0 = (wsel0 == 0) ? Qo : (wsel0 == 1) ? Ko : Vo;
        __bf16* outp1 = (wsel1 == 0) ? Qo : (wsel1 == 1) ? Ko : Vo;
        const int n0 = nT0 * 16 + lm, n1 = nT1 * 16 + lm;
#pragma unroll
        for (int r = 0; r < 8; ++r) {              // D tile: M = r + 8*lc, N = lm
            const size_t tokA = m0 + r + 8 * lc;
            const size_t tokB = tokA + 16;
            outp0[tokA * HID + n0] = (__bf16)acc00[r];
            outp1[tokA * HID + n1] = (__bf16)acc01[r];
            outp0[tokB * HID + n0] = (__bf16)acc10[r];
            outp1[tokB * HID + n1] = (__bf16)acc11[r];
        }
    }
}

// ---------------------------------------------------------------------------
// Kernel 3: per-token head-mixing "attention".  1 wave = 1 token.
// S[16x16] = (q[16x64] @ k^T[64x16]) / 8  -> softmax rows -> O = P @ v[16x64].
// ---------------------------------------------------------------------------
__global__ __launch_bounds__(256, 1) void attn(const __bf16* __restrict__ Q,
                                               const __bf16* __restrict__ K,
                                               const __bf16* __restrict__ V,
                                               __bf16* __restrict__ O) {
    __shared__ float  sS[8][256];
    __shared__ __bf16 sP[8][256];
    __shared__ __bf16 sV[8][1024];
    const int tid = threadIdx.x, w = tid >> 5, lane = tid & 31;
    const int lm = lane & 15, lc = lane >> 4;
    const size_t tok = (size_t)blockIdx.x * 8 + w;
    const __bf16* q = Q + tok * HID;
    const __bf16* k = K + tok * HID;
    const __bf16* v = V + tok * HID;

    // Stage v row (1024 bf16) to LDS for the transposed gather later.
    {
        const u32x4* vs = (const u32x4*)v;
        u32x4*       vd = (u32x4*)sV[w];
#pragma unroll
        for (int i = 0; i < 4; ++i) vd[lane + i * 32] = vs[lane + i * 32];
    }

    // S = q @ k^T  (A rows = heads, K = head_dim 64 -> two WMMA steps).
    f32x8 s = {};
#pragma unroll
    for (int kt = 0; kt < 2; ++kt) {
        const __bf16* ap = q + lm * 64 + kt * 32 + lc * 8;
        const bf16x16 a = load_frag_lds(ap, ap + 16);
        // B(d, t): col t = lm, K = lc*8 + j (+8 for j >= 8) of k-row t.
        const __bf16* bp = k + lm * 64 + kt * 32 + lc * 8;
        const bf16x16 b = load_frag_lds(bp, bp + 16);
        s = __builtin_amdgcn_wmma_f32_16x16x32_bf16(
            false, a, false, b, (short)0, s, false, false);
    }
#pragma unroll
    for (int r = 0; r < 8; ++r)
        sS[w][(r + 8 * lc) * 16 + lm] = s[r] * 0.125f;   // 1/sqrt(64)
    __syncthreads();

    // Row softmax (16 rows, lanes 0..15).
    if (lane < 16) {
        float rowv[16], m = -INFINITY;
#pragma unroll
        for (int j = 0; j < 16; ++j) { rowv[j] = sS[w][lane * 16 + j]; m = fmaxf(m, rowv[j]); }
        float sum = 0.f;
#pragma unroll
        for (int j = 0; j < 16; ++j) { rowv[j] = __expf(rowv[j] - m); sum += rowv[j]; }
        const float inv = 1.0f / sum;
#pragma unroll
        for (int j = 0; j < 16; ++j) sP[w][lane * 16 + j] = (__bf16)(rowv[j] * inv);
    }
    __syncthreads();

    // P as A-fragment with effective K = 16 (upper-K half zeroed).
    BF16Frag pf;
    pf.u[0] = *(const u32x4*)&sP[w][lm * 16 + lc * 8];   // K = lc*8 .. +8
    pf.u[1] = (u32x4)(0u);                               // K >= 16 -> 0
    const bf16x16 pa = pf.v;

    // O = P @ v : 4 N-tiles of 16 dims; B elements j >= 8 carry K >= 16 -> 0.
#pragma unroll
    for (int nt = 0; nt < 4; ++nt) {
        bf16x16 b;
#pragma unroll
        for (int j = 0; j < 16; ++j)
            b[j] = (j < 8) ? sV[w][(lc * 8 + j) * 64 + nt * 16 + lm] : (__bf16)0.0f;
        f32x8 o = {};
        o = __builtin_amdgcn_wmma_f32_16x16x32_bf16(
            false, pa, false, b, (short)0, o, false, false);
#pragma unroll
        for (int r = 0; r < 8; ++r)
            O[tok * HID + (size_t)(r + 8 * lc) * 64 + nt * 16 + lm] = (__bf16)o[r];
    }
}

// ---------------------------------------------------------------------------
// Kernel 4: output projection  Y = O @ Wo^T  (bf16 in, fp32 out to d_out).
// 1 block = 32 tokens; 64 N-tiles over 8 waves as 4 pairs, 2 M-subtiles.
// ---------------------------------------------------------------------------
__global__ __launch_bounds__(256, 1) void out_proj(const __bf16* __restrict__ Oin,
                                                   const __bf16* __restrict__ Wp,
                                                   float* __restrict__ Y) {
    __shared__ __bf16 sA[32 * LDSTRIDE];
    const int tid = threadIdx.x;
    const size_t m0 = (size_t)blockIdx.x * 32;
#pragma unroll
    for (int i = 0; i < 16; ++i) {                 // 4096 16-byte chunks
        const int c = tid + i * 256;
        const int e = c * 8, row = e >> 10, col = e & 1023;
        *(u32x4*)&sA[row * LDSTRIDE + col] = *(const u32x4*)(Oin + (m0 + row) * HID + col);
    }
    __syncthreads();

    const int w = tid >> 5, lane = tid & 31;
    const int lm = lane & 15, lc = lane >> 4;
    for (int i = 0; i < 4; ++i) {
        const int nT0 = w * 8 + i;
        const int nT1 = nT0 + 4;
        const __bf16* wp0 = Wp + ((size_t)nT0 * 32 * 32 + lane) * 16;
        const __bf16* wp1 = Wp + ((size_t)nT1 * 32 * 32 + lane) * 16;
        f32x8 acc00 = {}, acc01 = {}, acc10 = {}, acc11 = {};
#pragma unroll 2
        for (int kT = 0; kT < KTILES; ++kT) {
            const int ab0 = lm * LDSTRIDE + kT * 32 + lc * 8;
            const int ab1 = (lm + 16) * LDSTRIDE + kT * 32 + lc * 8;
            const bf16x16 a0 = load_frag_lds(&sA[ab0], &sA[ab0 + 16]);
            const bf16x16 a1 = load_frag_lds(&sA[ab1], &sA[ab1 + 16]);
            const bf16x16 b0 = load_frag_glb(wp0 + (size_t)kT * 512);
            const bf16x16 b1 = load_frag_glb(wp1 + (size_t)kT * 512);
            __builtin_prefetch(wp0 + (size_t)(kT + 4) * 512, 0, 3);
            __builtin_prefetch(wp1 + (size_t)(kT + 4) * 512, 0, 3);
            acc00 = __builtin_amdgcn_wmma_f32_16x16x32_bf16(
                false, a0, false, b0, (short)0, acc00, false, false);
            acc01 = __builtin_amdgcn_wmma_f32_16x16x32_bf16(
                false, a0, false, b1, (short)0, acc01, false, false);
            acc10 = __builtin_amdgcn_wmma_f32_16x16x32_bf16(
                false, a1, false, b0, (short)0, acc10, false, false);
            acc11 = __builtin_amdgcn_wmma_f32_16x16x32_bf16(
                false, a1, false, b1, (short)0, acc11, false, false);
        }
        const int n0 = nT0 * 16 + lm, n1 = nT1 * 16 + lm;
#pragma unroll
        for (int r = 0; r < 8; ++r) {
            const size_t tokA = m0 + r + 8 * lc;
            const size_t tokB = tokA + 16;
            Y[tokA * HID + n0] = acc00[r];
            Y[tokA * HID + n1] = acc01[r];
            Y[tokB * HID + n0] = acc10[r];
            Y[tokB * HID + n1] = acc11[r];
        }
    }
}

// ---------------------------------------------------------------------------
extern "C" void kernel_launch(void* const* d_in, const int* in_sizes, int n_in,
                              void* d_out, int out_size, void* d_ws, size_t ws_size,
                              hipStream_t stream) {
    const float* X  = (const float*)d_in[0];
    const float* Wq = (const float*)d_in[1];
    const float* Wk = (const float*)d_in[2];
    const float* Wv = (const float*)d_in[3];
    const float* Wo = (const float*)d_in[4];

    char* ws = (char*)d_ws;
    const size_t WPACK_BYTES = 4ull * HID * HID * 2;         // 8 MB
    const size_t MAT_BYTES   = (size_t)NTOK * HID * 2;       // 64 MB each
    __bf16* Wpack = (__bf16*)ws;
    __bf16* Qb    = (__bf16*)(ws + WPACK_BYTES);
    __bf16* Kb    = (__bf16*)(ws + WPACK_BYTES + 1 * MAT_BYTES);
    __bf16* Vb    = (__bf16*)(ws + WPACK_BYTES + 2 * MAT_BYTES);
    __bf16* Ob    = (__bf16*)(ws + WPACK_BYTES + 3 * MAT_BYTES);

    pack_w  <<<1024, 256, 0, stream>>>(Wq, Wk, Wv, Wo, Wpack);
    qkv_gemm<<<NTOK / 32, 256, 0, stream>>>(X, Wpack, Qb, Kb, Vb);
    attn    <<<NTOK / 8, 256, 0, stream>>>(Qb, Kb, Vb, Ob);
    out_proj<<<NTOK / 32, 256, 0, stream>>>(Ob, Wpack + 3ull * HID * HID, (float*)d_out);
}
